// EnhancedAttentionLayer_41077067219266
// MI455X (gfx1250) — matrix operand order repaired and checked by reading, not compiled, not run on hardware
//
#include <hip/hip_runtime.h>
#include <hip/hip_bf16.h>

typedef __attribute__((ext_vector_type(16))) _Float16 v16h;
typedef __attribute__((ext_vector_type(8)))  float    v8f;

#define WMMA_F16(A,B,C) __builtin_amdgcn_wmma_f32_16x16x32_f16(false,(A),false,(B),(short)0,(C),false,false)

// Problem constants (from setup_inputs: B=4, C=64, H=W=256)
static constexpr int NB = 4;
static constexpr int NC = 64;
static constexpr int NP = 65536;           // H*W pixels per batch

// Workspace layout (float offsets)
static constexpr int GX_OFF   = 0;         // [B][64][64] gram of x
static constexpr int GY_OFF   = 16384;     // [B][64][64] gram of y
static constexpr int SX_OFF   = 32768;     // [B][64] channel sums of x
static constexpr int SY_OFF   = 33024;     // [B][64] channel sums of y
static constexpr int ATX_OFF  = 33280;     // [B][64][8] attn(+I) for x
static constexpr int ATY_OFF  = 35328;     // [B][64][8] attn(+I) for y
static constexpr int ST_OFF   = 37376;     // [B][2] (max, 1/sumexp)
static constexpr int LOG_OFF  = 40960;     // [B][65536] logits
static constexpr int ZERO_N   = 33280;     // accumulators that must start at 0

static constexpr int CHUNK = 256;          // pixels per gram workgroup

// ---------------------------------------------------------------------------
__global__ void zero_kernel(float* ws, int n) {
    int i = blockIdx.x * 256 + threadIdx.x;
    if (i < n) ws[i] = 0.f;
}

// ---------------------------------------------------------------------------
// Pass 1: per-batch input Gram G = X * X^T (K = pixels) via WMMA, + channel sums.
__global__ __launch_bounds__(256) void gram_kernel(const float* __restrict__ x,
                                                   const float* __restrict__ y,
                                                   float* __restrict__ ws) {
    __shared__ _Float16 lds[NC][CHUNK + 8];   // +8 halfs pad: conflict-free column reads
    const int inp = blockIdx.y;
    const float* src = inp ? y : x;
    const int b  = blockIdx.x >> 8;           // 256 chunks per batch
    const int p0 = (blockIdx.x & 255) * CHUNK;
    const float* base = src + (size_t)b * NC * NP + p0;
    const int tid = threadIdx.x;

    // Stage chunk to LDS as f16 (coalesced float4 reads)
    for (int i = tid; i < NC * (CHUNK / 4); i += 256) {
        int c = i >> 6;
        int p = (i & 63) * 4;
        float4 v = *(const float4*)(base + (size_t)c * NP + p);
        lds[c][p + 0] = (_Float16)v.x; lds[c][p + 1] = (_Float16)v.y;
        lds[c][p + 2] = (_Float16)v.z; lds[c][p + 3] = (_Float16)v.w;
    }
    __syncthreads();

    const int w = tid >> 5, lane = tid & 31, hi = lane >> 4, lm = lane & 15;
    float* G = ws + (inp ? GY_OFF : GX_OFF) + b * 4096;

    #pragma unroll
    for (int pp = 0; pp < 2; ++pp) {          // 16 (ct,dt) tile pairs over 8 waves
        int pid = w * 2 + pp, ct = pid >> 2, dt = pid & 3;
        v8f acc = {0.f, 0.f, 0.f, 0.f, 0.f, 0.f, 0.f, 0.f};
        for (int k0 = 0; k0 < CHUNK; k0 += 32) {
            v16h a, bf;
            int ca = ct * 16 + lm;
            #pragma unroll
            for (int j = 0; j < 8; ++j) {     // A layout: k = e + (e>=8)*8 + hi*8
                int px = k0 + ((j < 4) ? 2 * j : 16 + 2 * (j - 4)) + hi * 8;
                a[2 * j]     = lds[ca][px];
                a[2 * j + 1] = lds[ca][px + 1];
            }
            int cb = dt * 16 + lm;
            int pbx = k0 + hi * 8;            // B layout: k = e + (e>=8)*8 + hi*8
            #pragma unroll
            for (int e = 0; e < 16; ++e) bf[e] = lds[cb][pbx + e + ((e >= 8) ? 8 : 0)];
            acc = WMMA_F16(a, bf, acc);
        }
        #pragma unroll
        for (int r = 0; r < 8; ++r)           // D layout: m = r + 8*hi, n = lm
            atomicAdd(&G[(ct * 16 + r + 8 * hi) * 64 + dt * 16 + lm], acc[r]);
    }

    // Per-channel sums (for bias generality; biases are zero in this harness)
    float* sv = ws + (inp ? SY_OFF : SX_OFF) + b * 64;
    int c = tid & 63;
    float s = 0.f;
    for (int p = tid >> 6; p < CHUNK; p += 4) s += (float)lds[c][p];
    atomicAdd(&sv[c], s);
}

// ---------------------------------------------------------------------------
// Pass 1.5: derive per-head softmaxed attention (+identity) from the Grams.
__global__ __launch_bounds__(256) void stats_kernel(const float* __restrict__ wq,
                                                    const float* __restrict__ bq,
                                                    const float* __restrict__ wk,
                                                    const float* __restrict__ bk,
                                                    const float* __restrict__ temp,
                                                    float* __restrict__ ws) {
    __shared__ float Tq[64][64], Tk[64][64];
    __shared__ float Sq[64], Sk[64], wqs[64], wks[64];
    __shared__ float att[64][8];
    const int b = blockIdx.x, inp = blockIdx.y;
    const float* G  = ws + (inp ? GY_OFF : GX_OFF) + b * 4096;
    const float* sv = ws + (inp ? SY_OFF : SX_OFF) + b * 64;
    const int tid = threadIdx.x;

    // T = W * G
    for (int e = tid; e < 4096; e += 256) {
        int c = e >> 6, d = e & 63;
        float aq = 0.f, ak = 0.f;
        for (int i = 0; i < 64; ++i) {
            float g = G[i * 64 + d];
            aq += wq[c * 64 + i] * g;
            ak += wk[c * 64 + i] * g;
        }
        Tq[c][d] = aq; Tk[c][d] = ak;
    }
    __syncthreads();

    if (tid < 64) {   // ||q_c||^2 = (Wq G Wq^T)[c,c] + bias terms
        int c = tid;
        float sq = 0.f, sk = 0.f, qs = 0.f, ks = 0.f;
        for (int d = 0; d < 64; ++d) {
            sq += Tq[c][d] * wq[c * 64 + d];
            sk += Tk[c][d] * wk[c * 64 + d];
            qs += wq[c * 64 + d] * sv[d];
            ks += wk[c * 64 + d] * sv[d];
        }
        Sq[c] = sq + 2.f * bq[c] * qs + 65536.f * bq[c] * bq[c];
        Sk[c] = sk + 2.f * bk[c] * ks + 65536.f * bk[c] * bk[c];
        wqs[c] = qs; wks[c] = ks;
    }
    __syncthreads();

    for (int e = tid; e < 512; e += 256) {   // within-head q.k^T Gram
        int c = e >> 3, d = e & 7;
        int dg = (c & ~7) + d;
        float v = 0.f;
        for (int j = 0; j < 64; ++j) v += Tq[c][j] * wk[dg * 64 + j];
        v += bq[c] * wks[dg] + bk[dg] * wqs[c] + 65536.f * bq[c] * bk[dg];
        att[c][d] = v;
    }
    __syncthreads();

    if (tid < 64) {   // softmax row + fold identity (attn@v + v = (attn+I)@v)
        int c = tid, h = c >> 3;
        float nq = fmaxf(sqrtf(fmaxf(Sq[c], 0.f)), 1e-12f);
        float t  = temp[h];
        float row[8], mx = -3.4e38f;
        #pragma unroll
        for (int d = 0; d < 8; ++d) {
            float nk = fmaxf(sqrtf(fmaxf(Sk[(c & ~7) + d], 0.f)), 1e-12f);
            row[d] = att[c][d] / (nq * nk) * t;
            mx = fmaxf(mx, row[d]);
        }
        float sum = 0.f;
        #pragma unroll
        for (int d = 0; d < 8; ++d) { row[d] = expf(row[d] - mx); sum += row[d]; }
        float inv = 1.f / sum;
        float* dst = ws + (inp ? ATY_OFF : ATX_OFF) + b * 512 + c * 8;
        #pragma unroll
        for (int d = 0; d < 8; ++d)
            dst[d] = row[d] * inv + ((d == (c & 7)) ? 1.f : 0.f);
    }
}

// ---------------------------------------------------------------------------
// Pass 2: fused v-proj -> attention apply -> residual -> ch/y proj -> tanh -> logit.
__global__ __launch_bounds__(256) void fused_kernel(const float* __restrict__ x,
                                                    const float* __restrict__ y,
                                                    const float* __restrict__ wvm,
                                                    const float* __restrict__ bv,
                                                    const float* __restrict__ wchm,
                                                    const float* __restrict__ bch,
                                                    const float* __restrict__ wym,
                                                    const float* __restrict__ waw,
                                                    const float* __restrict__ baw,
                                                    float* __restrict__ ws) {
    __shared__ __align__(32) _Float16 tr[8][2][16][80];  // per-wave D->B transpose, padded
    __shared__ float attA[2][64][8];
    __shared__ float sBv[64], sBch[64], sWaw[64];

    const int tid = threadIdx.x;
    const int b  = blockIdx.x >> 9;            // 512 blocks per batch (128 px each)
    const int p0 = (blockIdx.x & 511) * 128;

    for (int i = tid; i < 512; i += 256) attA[0][i >> 3][i & 7] = ws[ATX_OFF + b * 512 + i];
    for (int i = tid; i < 512; i += 256) attA[1][i >> 3][i & 7] = ws[ATY_OFF + b * 512 + i];
    if (tid < 64) { sBv[tid] = bv[tid]; sBch[tid] = bch[tid]; sWaw[tid] = waw[tid]; }
    __syncthreads();

    const int w = tid >> 5, lane = tid & 31, hi = lane >> 4, lm = lane & 15;
    const int pb = p0 + w * 16;
    const float* xb = x + (size_t)b * NC * NP;
    const float* yb = y + (size_t)b * NC * NP;

    // Wv A-fragments (held in registers; ACH/AY loaded later to bound pressure)
    v16h AV[4][2];
    #pragma unroll
    for (int ct = 0; ct < 4; ++ct)
        #pragma unroll
        for (int kt = 0; kt < 2; ++kt) {
            int o = ct * 16 + lm;
            #pragma unroll
            for (int j = 0; j < 8; ++j) {
                int i = kt * 32 + hi * 16 + 2 * j;
                AV[ct][kt][2 * j]     = (_Float16)wvm[o * 64 + i];
                AV[ct][kt][2 * j + 1] = (_Float16)wvm[o * 64 + i + 1];
            }
        }

    // B-fragments of x and y (pixels are N, channels are K)
    v16h BX[2], BY[2];
    #pragma unroll
    for (int kt = 0; kt < 2; ++kt)
        #pragma unroll
        for (int e = 0; e < 16; ++e) {
            int c = kt * 32 + hi * 16 + e;
            BX[kt][e] = (_Float16)xb[(size_t)c * NP + pb + lm];
            BY[kt][e] = (_Float16)yb[(size_t)c * NP + pb + lm];
        }

    // v = Wv * input  (16 WMMAs per wave)
    v8f VX[4], VY[4];
    #pragma unroll
    for (int ct = 0; ct < 4; ++ct) {
        v8f a = {0.f,0.f,0.f,0.f,0.f,0.f,0.f,0.f};
        a = WMMA_F16(AV[ct][0], BX[0], a);
        a = WMMA_F16(AV[ct][1], BX[1], a);
        VX[ct] = a;
        v8f c2 = {0.f,0.f,0.f,0.f,0.f,0.f,0.f,0.f};
        c2 = WMMA_F16(AV[ct][0], BY[0], c2);
        c2 = WMMA_F16(AV[ct][1], BY[1], c2);
        VY[ct] = c2;
    }

    // Per-lane 8x8 attention apply (lane holds one full head) + residual
    float XA[4][8], YA[4][8];
    #pragma unroll
    for (int ct = 0; ct < 4; ++ct) {
        int h = 2 * ct + hi;
        #pragma unroll
        for (int r = 0; r < 8; ++r) {
            int c = ct * 16 + hi * 8 + r;
            float ax = xb[(size_t)c * NP + pb + lm];
            float ay = yb[(size_t)c * NP + pb + lm];
            #pragma unroll
            for (int d = 0; d < 8; ++d) {
                float vxx = VX[ct][d] + sBv[h * 8 + d];
                float vyy = VY[ct][d] + sBv[h * 8 + d];
                ax += attA[0][h * 8 + r][d] * vxx;
                ay += attA[1][h * 8 + r][d] * vyy;
            }
            XA[ct][r] = ax; YA[ct][r] = ay;
        }
    }

    // D-layout -> B-layout relayout through LDS
    #pragma unroll
    for (int ct = 0; ct < 4; ++ct)
        #pragma unroll
        for (int r = 0; r < 8; ++r) {
            tr[w][0][lm][ct * 16 + hi * 8 + r] = (_Float16)XA[ct][r];
            tr[w][1][lm][ct * 16 + hi * 8 + r] = (_Float16)YA[ct][r];
        }
    __syncthreads();

    v16h BXA[2], BYA[2];
    #pragma unroll
    for (int kt = 0; kt < 2; ++kt) {
        BXA[kt] = *reinterpret_cast<const v16h*>(&tr[w][0][lm][kt * 32 + hi * 16]);
        BYA[kt] = *reinterpret_cast<const v16h*>(&tr[w][1][lm][kt * 32 + hi * 16]);
    }

    // W_ch / W_y A-fragments
    v16h ACH[4][2], AY[4][2];
    #pragma unroll
    for (int ct = 0; ct < 4; ++ct)
        #pragma unroll
        for (int kt = 0; kt < 2; ++kt) {
            int o = ct * 16 + lm;
            #pragma unroll
            for (int j = 0; j < 8; ++j) {
                int i = kt * 32 + hi * 16 + 2 * j;
                ACH[ct][kt][2 * j]     = (_Float16)wchm[o * 64 + i];
                ACH[ct][kt][2 * j + 1] = (_Float16)wchm[o * 64 + i + 1];
                AY[ct][kt][2 * j]      = (_Float16)wym[o * 64 + i];
                AY[ct][kt][2 * j + 1]  = (_Float16)wym[o * 64 + i + 1];
            }
        }

    // ch = Wch*x_attn + Wy*y_attn (16 WMMAs), then tanh + logit reduce
    float lg = 0.f;
    #pragma unroll
    for (int ct = 0; ct < 4; ++ct) {
        v8f a = {0.f,0.f,0.f,0.f,0.f,0.f,0.f,0.f};
        a = WMMA_F16(ACH[ct][0], BXA[0], a);
        a = WMMA_F16(ACH[ct][1], BXA[1], a);
        a = WMMA_F16(AY[ct][0],  BYA[0], a);
        a = WMMA_F16(AY[ct][1],  BYA[1], a);
        #pragma unroll
        for (int r = 0; r < 8; ++r) {
            int c = ct * 16 + hi * 8 + r;
            lg += sWaw[c] * tanhf(a[r] + sBch[c]);
        }
    }
    lg += __shfl_xor(lg, 16, 32);              // combine the two channel halves
    if (hi == 0) ws[LOG_OFF + b * NP + pb + lm] = lg + baw[0];
}

// ---------------------------------------------------------------------------
// Pass 3: per-batch softmax statistics over 65536 logits.
__global__ __launch_bounds__(256) void reduce_kernel(float* __restrict__ ws) {
    __shared__ float red[256];
    const int b = blockIdx.x, tid = threadIdx.x;
    const float* lg = ws + LOG_OFF + (size_t)b * NP;
    float m = -3.4e38f;
    for (int i = tid; i < NP; i += 256) m = fmaxf(m, lg[i]);
    red[tid] = m; __syncthreads();
    for (int s = 128; s > 0; s >>= 1) { if (tid < s) red[tid] = fmaxf(red[tid], red[tid + s]); __syncthreads(); }
    float mx = red[0]; __syncthreads();
    float s = 0.f;
    for (int i = tid; i < NP; i += 256) s += expf(lg[i] - mx);
    red[tid] = s; __syncthreads();
    for (int st = 128; st > 0; st >>= 1) { if (tid < st) red[tid] += red[tid + st]; __syncthreads(); }
    if (tid == 0) { ws[ST_OFF + b * 2] = mx; ws[ST_OFF + b * 2 + 1] = 1.f / red[0]; }
}

// ---------------------------------------------------------------------------
// Pass 4: result = input * (1 + softmax_weight), both tensors.
__global__ __launch_bounds__(256) void scale_kernel(const float* __restrict__ x,
                                                    const float* __restrict__ y,
                                                    const float* __restrict__ ws,
                                                    float* __restrict__ out) {
    const int idx = blockIdx.x * 256 + threadIdx.x;      // 65536 threads
    const int b = idx >> 14;
    const int n = (idx & 16383) * 4;
    const float mx  = ws[ST_OFF + b * 2];
    const float inv = ws[ST_OFF + b * 2 + 1];
    float4 lg = *(const float4*)(ws + LOG_OFF + (size_t)b * NP + n);
    float4 f;
    f.x = 1.f + expf(lg.x - mx) * inv;
    f.y = 1.f + expf(lg.y - mx) * inv;
    f.z = 1.f + expf(lg.z - mx) * inv;
    f.w = 1.f + expf(lg.w - mx) * inv;
    const float* xb = x + (size_t)b * NC * NP + n;
    const float* yb = y + (size_t)b * NC * NP + n;
    float* o1 = out + (size_t)b * NC * NP + n;
    float* o2 = out + (size_t)NB * NC * NP + (size_t)b * NC * NP + n;
    for (int c = 0; c < NC; ++c) {
        if (c + 4 < NC) {
            __builtin_prefetch(xb + (size_t)(c + 4) * NP, 0, 0);
            __builtin_prefetch(yb + (size_t)(c + 4) * NP, 0, 0);
        }
        float4 xv = *(const float4*)(xb + (size_t)c * NP);
        float4 yv = *(const float4*)(yb + (size_t)c * NP);
        float4 r1 = {xv.x * f.x, xv.y * f.y, xv.z * f.z, xv.w * f.w};
        float4 r2 = {yv.x * f.x, yv.y * f.y, yv.z * f.z, yv.w * f.w};
        *(float4*)(o1 + (size_t)c * NP) = r1;
        *(float4*)(o2 + (size_t)c * NP) = r2;
    }
}

// ---------------------------------------------------------------------------
extern "C" void kernel_launch(void* const* d_in, const int* in_sizes, int n_in,
                              void* d_out, int out_size, void* d_ws, size_t ws_size,
                              hipStream_t stream) {
    const float* x    = (const float*)d_in[0];
    const float* y    = (const float*)d_in[1];
    const float* wq   = (const float*)d_in[2];
    const float* bq   = (const float*)d_in[3];
    const float* wk   = (const float*)d_in[4];
    const float* bk   = (const float*)d_in[5];
    const float* wv   = (const float*)d_in[6];
    const float* bv   = (const float*)d_in[7];
    const float* temp = (const float*)d_in[8];
    const float* wch  = (const float*)d_in[9];
    const float* bch  = (const float*)d_in[10];
    const float* wy   = (const float*)d_in[11];
    const float* waw  = (const float*)d_in[12];
    const float* baw  = (const float*)d_in[13];
    float* ws  = (float*)d_ws;
    float* out = (float*)d_out;
    (void)in_sizes; (void)n_in; (void)out_size; (void)ws_size;

    zero_kernel<<<(ZERO_N + 255) / 256, 256, 0, stream>>>(ws, ZERO_N);
    gram_kernel<<<dim3(NB * (NP / CHUNK), 2), 256, 0, stream>>>(x, y, ws);
    stats_kernel<<<dim3(NB, 2), 256, 0, stream>>>(wq, bq, wk, bk, temp, ws);
    fused_kernel<<<NB * (NP / 128), 256, 0, stream>>>(x, y, wv, bv, wch, bch, wy, waw, baw, ws);
    reduce_kernel<<<NB, 256, 0, stream>>>(ws);
    scale_kernel<<<NB * NP / 4 / 256, 256, 0, stream>>>(x, y, ws, out);
}
